// GatedCrossAttention_41051297415800
// MI455X (gfx1250) — matrix operand order repaired and checked by reading, not compiled, not run on hardware
//
#include <hip/hip_runtime.h>
#include <hip/hip_bf16.h>
#include <cstdint>

// ---------------- CDNA5 WMMA vector types ----------------
typedef __attribute__((ext_vector_type(16))) __bf16 v16bf;
typedef __attribute__((ext_vector_type(8)))  __bf16 v8bf;
typedef __attribute__((ext_vector_type(8)))  float  v8f;

static constexpr int GB  = 4;     // batch
static constexpr int GTT = 2048;  // text tokens
static constexpr int GTA = 1500;  // audio tokens
static constexpr int GD  = 2048;  // hidden
static constexpr int GH  = 16;    // heads
static constexpr int GDH = 128;   // head dim

// ---------------- f32 -> bf16 convert (grid-stride) ----------------
__global__ void cvt_f32_bf16_kernel(const float* __restrict__ x,
                                    __bf16* __restrict__ y, size_t n) {
  size_t i = (size_t)blockIdx.x * blockDim.x + threadIdx.x;
  size_t stride = (size_t)gridDim.x * blockDim.x;
  for (; i < n; i += stride) y[i] = (__bf16)x[i];
}

// ---------------- LayerNorm row kernel (audio_kv -> bf16) ----------------
__global__ __launch_bounds__(256)
void ln_bf16_kernel(const float* __restrict__ x, const float* __restrict__ gamma,
                    const float* __restrict__ beta, __bf16* __restrict__ y, int D) {
  const int row = blockIdx.x;
  const float* xr = x + (size_t)row * D;
  float lsum = 0.f, lsq = 0.f;
  for (int i = threadIdx.x; i < D; i += blockDim.x) {
    float v = xr[i];
    lsum += v; lsq += v * v;
  }
  __shared__ float s1[256], s2[256];
  s1[threadIdx.x] = lsum; s2[threadIdx.x] = lsq;
  __syncthreads();
  for (int off = 128; off > 0; off >>= 1) {
    if (threadIdx.x < off) {
      s1[threadIdx.x] += s1[threadIdx.x + off];
      s2[threadIdx.x] += s2[threadIdx.x + off];
    }
    __syncthreads();
  }
  const float mu  = s1[0] / D;
  const float var = s2[0] / D - mu * mu;
  const float rstd = rsqrtf(var + 1e-5f);
  for (int i = threadIdx.x; i < D; i += blockDim.x)
    y[(size_t)row * D + i] = (__bf16)((xr[i] - mu) * rstd * gamma[i] + beta[i]);
}

// ---------------- WMMA GEMM: C[M,N] = A[M,K] * W[N,K]^T ----------------
// A-tile (16 x 128 per k-superstep) staged into LDS with
// global_load_async_to_lds_b128 (double-buffered, ASYNCcnt-fenced), shared by
// all 8 waves; W fragments stream from global (whole weight is L2-resident).
// MODE 0: bf16 scatter to [B, H, Tdim, Dh]   (Q and K projections)
// MODE 2: bf16 scatter to [B, H, Dh, Tdim]   (V projection, transposed)
// MODE 3: f32 epilogue: out = resid + tanh(gate)*acc (row-major)
template <int MODE>
__global__ __launch_bounds__(256)
void wmma_gemm_kernel(const __bf16* __restrict__ A, const __bf16* __restrict__ W,
                      void* __restrict__ outp, int M, int N, int K, int Tdim,
                      const float* __restrict__ resid, const float* __restrict__ gate) {
  __shared__ __align__(16) __bf16 sA[2][16 * 128];

  const int tid  = threadIdx.x;
  const int lane = tid & 31;
  const int wave = tid >> 5;
  const int m0 = blockIdx.x * 16;
  const int n0 = blockIdx.y * 128 + wave * 16;

  const int arow   = lane & 15;          // A: row this lane owns
  const int koffA  = (lane >> 4) << 4;   // A: +16 K-offset for upper half-wave
  const int kbaseB = (lane >> 4) << 4;   // B: +16 K-offset for upper half-wave
  const int wrow   = n0 + (lane & 15);   // B: W row (output column)
  const __bf16* Wp = W + (size_t)wrow * K;

  // staging mapping: thread t copies 16 B (row t/16, 16B-group t%16)
  const int sr  = tid >> 4;
  const int scg = tid & 15;
  const __bf16* Asrc = A + (size_t)(m0 + sr) * K + scg * 8;
  const bool srow_ok = (m0 + sr) < M;

  auto stage = [&](int buf, int ks) {
    if (srow_ok) {
      unsigned lds_off = (unsigned)(uintptr_t)&sA[buf][sr * 128 + scg * 8];
      unsigned long long gaddr = (unsigned long long)(uintptr_t)(Asrc + ks);
      asm volatile("global_load_async_to_lds_b128 %0, %1, off"
                   :: "v"(lds_off), "v"(gaddr) : "memory");
    } else {
      v8bf z = {};
      *(v8bf*)&sA[buf][sr * 128 + scg * 8] = z;
    }
  };

  v8f acc = {};
  stage(0, 0);
  const int nsteps = K >> 7;     // K / 128
  for (int s = 0; s < nsteps; ++s) {
    asm volatile("s_wait_asynccnt 0" ::: "memory");
    __syncthreads();
    if (s + 1 < nsteps) stage((s + 1) & 1, (s + 1) << 7);

    const __bf16* sArow = &sA[s & 1][arow * 128];
    const int ks = s << 7;
    __builtin_prefetch(Wp + ks + 128, 0, 1);   // next W k-slab -> global_prefetch
#pragma unroll
    for (int kc = 0; kc < 4; ++kc) {
      const int k0 = kc * 32;
      // A fragment from LDS: two aligned 16B runs (contiguous 16-k per half-wave)
      v8bf lo = *(const v8bf*)(sArow + k0 + koffA);
      v8bf hi = *(const v8bf*)(sArow + k0 + 8 + koffA);
      v16bf af;
#pragma unroll
      for (int h = 0; h < 8; ++h) { af[h] = lo[h]; af[h + 8] = hi[h]; }
      // B fragment: 16 contiguous bf16 (32B aligned)
      v16bf bv = *(const v16bf*)(Wp + ks + k0 + kbaseB);
      acc = __builtin_amdgcn_wmma_f32_16x16x32_bf16(false, af, false, bv,
                                                    (short)0, acc, false, false);
    }
  }

  const float tg = (MODE == 3) ? tanhf(gate[0]) : 0.f;
#pragma unroll
  for (int v = 0; v < 8; ++v) {
    const int m = m0 + v + ((lane >> 4) << 3);  // ISA C-layout rows
    const int n = n0 + (lane & 15);
    if (m >= M) continue;
    const float val = acc[v];
    if constexpr (MODE == 3) {
      float* O = (float*)outp;
      const size_t idx = (size_t)m * N + n;
      O[idx] = resid[idx] + tg * val;
    } else {
      __bf16* O = (__bf16*)outp;
      const int b = m / Tdim, t = m % Tdim;
      const int h = n >> 7, d = n & 127;
      size_t idx;
      if constexpr (MODE == 0)
        idx = (((size_t)(b * GH + h) * Tdim) + t) * GDH + d;   // [B,H,T,Dh]
      else
        idx = (((size_t)(b * GH + h) * GDH) + d) * Tdim + t;   // [B,H,Dh,T]
      O[idx] = (__bf16)val;
    }
  }
}

// ---------------- Flash cross-attention: one wave per 16-query tile ----------------
__global__ __launch_bounds__(32)
void flash_attn_kernel(const __bf16* __restrict__ Qh, const __bf16* __restrict__ Kh,
                       const __bf16* __restrict__ Vt, const unsigned char* __restrict__ amask,
                       __bf16* __restrict__ Oout, int Tt, int Ta) {
  const int lane = threadIdx.x & 31;
  const int qt = blockIdx.x;            // 16-query tile index
  const int bh = blockIdx.y;            // b*H + h
  const int b = bh / GH, hh = bh % GH;
  const int hgrp = lane >> 4;           // half-wave select
  const int arow = lane & 15;
  const int koffA = hgrp << 4;
  const int kbaseB = hgrp << 4;
  const float scale = 0.08838834764831845f; // 1/sqrt(128)

  __shared__ __align__(32) float sP[16 * 32];

  // Preload Q tile as 4 A-fragments (Dh=128 = 4 x K32), vectorized 16B loads
  v16bf qa[4];
  const __bf16* Qp = Qh + ((size_t)bh * Tt + (size_t)qt * 16 + arow) * GDH;
#pragma unroll
  for (int c = 0; c < 4; ++c) {
    v8bf lo = *(const v8bf*)(Qp + c * 32 + koffA);
    v8bf hi = *(const v8bf*)(Qp + c * 32 + 8 + koffA);
#pragma unroll
    for (int h = 0; h < 8; ++h) { qa[c][h] = lo[h]; qa[c][h + 8] = hi[h]; }
  }

  const unsigned char* maskb = amask + (size_t)b * Ta;

  v8f o[8];
  const v8f zero8 = {};
#pragma unroll
  for (int t = 0; t < 8; ++t) o[t] = zero8;
  float mrow[8], lrow[8];
#pragma unroll
  for (int v = 0; v < 8; ++v) { mrow[v] = -1e30f; lrow[v] = 0.f; }

  for (int s0 = 0; s0 < Ta; s0 += 32) {
    // ---- branch-free, clamped mask bytes for both 16-key tiles ----
    const int key0  = s0 + arow;
    const int key1  = s0 + 16 + arow;
    const int key0c = (key0 < Ta) ? key0 : (Ta - 1);
    const int key1c = (key1 < Ta) ? key1 : (Ta - 1);
    const unsigned char mb0 = maskb[key0c];   // unconditional loads (clamped)
    const unsigned char mb1 = maskb[key1c];
    const bool kvt[2] = { (key0 < Ta) && (mb0 != 0),
                          (key1 < Ta) && (mb1 != 0) };
    const int keyct[2] = { key0c, key1c };

    // ---- scores for two 16-key tiles ----
    v8f sc[2];
#pragma unroll
    for (int nt = 0; nt < 2; ++nt) {
      const __bf16* Kp = Kh + ((size_t)bh * Ta + keyct[nt]) * GDH;
      const bool kv = kvt[nt];
      v8f acc = zero8;
#pragma unroll
      for (int c = 0; c < 4; ++c) {
        // invalid lanes compute garbage scores that are overwritten below
        v16bf kb = *(const v16bf*)(Kp + c * 32 + kbaseB);
        acc = __builtin_amdgcn_wmma_f32_16x16x32_bf16(false, qa[c], false, kb,
                                                      (short)0, acc, false, false);
      }
#pragma unroll
      for (int v = 0; v < 8; ++v) acc[v] = kv ? acc[v] * scale : -1e30f;
      sc[nt] = acc;
    }

    // ---- online softmax: row max over 16-lane half-waves ----
    float mnew[8], fs[8];
#pragma unroll
    for (int v = 0; v < 8; ++v) {
      float x = fmaxf(sc[0][v], sc[1][v]);
#pragma unroll
      for (int off = 1; off < 16; off <<= 1) x = fmaxf(x, __shfl_xor(x, off, 32));
      mnew[v] = fmaxf(mrow[v], x);
      fs[v] = __expf(mrow[v] - mnew[v]);
      mrow[v] = mnew[v];
    }
#pragma unroll
    for (int nt = 0; nt < 2; ++nt)
#pragma unroll
      for (int v = 0; v < 8; ++v) sc[nt][v] = __expf(sc[nt][v] - mnew[v]);
#pragma unroll
    for (int v = 0; v < 8; ++v) {
      float rs = sc[0][v] + sc[1][v];
#pragma unroll
      for (int off = 1; off < 16; off <<= 1) rs += __shfl_xor(rs, off, 32);
      lrow[v] = lrow[v] * fs[v] + rs;
    }
#pragma unroll
    for (int t = 0; t < 8; ++t)
#pragma unroll
      for (int v = 0; v < 8; ++v) o[t][v] *= fs[v];

    // ---- re-swizzle P: C-layout -> A-layout via LDS bounce ----
#pragma unroll
    for (int nt = 0; nt < 2; ++nt)
#pragma unroll
      for (int v = 0; v < 8; ++v)
        sP[(v + (hgrp << 3)) * 32 + nt * 16 + arow] = sc[nt][v];
    v8f p0 = *(const v8f*)&sP[arow * 32 + koffA];        // 32B-aligned ds_load
    v8f p1 = *(const v8f*)&sP[arow * 32 + 8 + koffA];
    v16bf pa;
#pragma unroll
    for (int h = 0; h < 8; ++h) { pa[h] = (__bf16)p0[h]; pa[h + 8] = (__bf16)p1[h]; }

    // ---- O += P * V (8 head-dim tiles) ----
    const bool full = (s0 + 32) <= Ta;
#pragma unroll
    for (int dt = 0; dt < 8; ++dt) {
      const int d = dt * 16 + arow;
      const __bf16* Vp = Vt + ((size_t)bh * GDH + d) * Ta + s0 + kbaseB;
      v16bf vb;
      if (full) {
        vb = *(const v16bf*)Vp;                          // 32B contiguous
      } else {
#pragma unroll
        for (int h = 0; h < 16; ++h) {
          const int key = s0 + kbaseB + h;
          const int idx = (key < Ta) ? h : (Ta - 1 - s0 - kbaseB);  // clamp; P==0 kills it
          vb[h] = Vp[idx];
        }
      }
      o[dt] = __builtin_amdgcn_wmma_f32_16x16x32_bf16(false, pa, false, vb,
                                                      (short)0, o[dt], false, false);
    }
  }

  // ---- normalize and store [B,Tt,H,Dh] bf16 ----
#pragma unroll
  for (int dt = 0; dt < 8; ++dt)
#pragma unroll
    for (int v = 0; v < 8; ++v) {
      const int q = qt * 16 + v + (hgrp << 3);
      const int d = dt * 16 + arow;
      Oout[(((size_t)b * Tt + q) * GH + hh) * GDH + d] = (__bf16)(o[dt][v] / lrow[v]);
    }
}

// ---------------- host launcher ----------------
extern "C" void kernel_launch(void* const* d_in, const int* in_sizes, int n_in,
                              void* d_out, int out_size, void* d_ws, size_t ws_size,
                              hipStream_t stream) {
  (void)in_sizes; (void)n_in; (void)out_size; (void)ws_size;
  const float* hs    = (const float*)d_in[0];
  const float* akv   = (const float*)d_in[1];
  const unsigned char* amask = (const unsigned char*)d_in[2];
  const float* Wq    = (const float*)d_in[3];
  const float* Wk    = (const float*)d_in[4];
  const float* Wv    = (const float*)d_in[5];
  const float* Wo    = (const float*)d_in[6];
  const float* gamma = (const float*)d_in[7];
  const float* beta  = (const float*)d_in[8];
  const float* gate  = (const float*)d_in[9];

  const int MT = GB * GTT;   // 8192 text rows
  const int MA = GB * GTA;   // 6000 audio rows

  char* ws = (char*)d_ws;
  auto carve = [&](size_t bytes) -> void* {
    void* p = (void*)ws;
    ws += (bytes + 255) & ~(size_t)255;
    return p;
  };
  __bf16* wq_b = (__bf16*)carve((size_t)GD * GD * 2);
  __bf16* wk_b = (__bf16*)carve((size_t)GD * GD * 2);
  __bf16* wv_b = (__bf16*)carve((size_t)GD * GD * 2);
  __bf16* wo_b = (__bf16*)carve((size_t)GD * GD * 2);
  __bf16* hs_b = (__bf16*)carve((size_t)MT * GD * 2);
  __bf16* a_b  = (__bf16*)carve((size_t)MA * GD * 2);
  __bf16* q_h  = (__bf16*)carve((size_t)MT * GD * 2);   // [B,H,Tt,Dh]
  __bf16* k_h  = (__bf16*)carve((size_t)MA * GD * 2);   // [B,H,Ta,Dh]
  __bf16* v_t  = (__bf16*)carve((size_t)MA * GD * 2);   // [B,H,Dh,Ta]
  __bf16* ao   = (__bf16*)carve((size_t)MT * GD * 2);   // [B,Tt,H*Dh]

  // 1) weight + activation conversions
  cvt_f32_bf16_kernel<<<4096, 256, 0, stream>>>(Wq, wq_b, (size_t)GD * GD);
  cvt_f32_bf16_kernel<<<4096, 256, 0, stream>>>(Wk, wk_b, (size_t)GD * GD);
  cvt_f32_bf16_kernel<<<4096, 256, 0, stream>>>(Wv, wv_b, (size_t)GD * GD);
  cvt_f32_bf16_kernel<<<4096, 256, 0, stream>>>(Wo, wo_b, (size_t)GD * GD);
  cvt_f32_bf16_kernel<<<8192, 256, 0, stream>>>(hs, hs_b, (size_t)MT * GD);

  // 2) LayerNorm audio_kv
  ln_bf16_kernel<<<MA, 256, 0, stream>>>(akv, gamma, beta, a_b, GD);

  // 3) projections
  wmma_gemm_kernel<0><<<dim3(MT / 16, GD / 128), 256, 0, stream>>>(
      hs_b, wq_b, (void*)q_h, MT, GD, GD, GTT, nullptr, nullptr);
  wmma_gemm_kernel<0><<<dim3(MA / 16, GD / 128), 256, 0, stream>>>(
      a_b, wk_b, (void*)k_h, MA, GD, GD, GTA, nullptr, nullptr);
  wmma_gemm_kernel<2><<<dim3(MA / 16, GD / 128), 256, 0, stream>>>(
      a_b, wv_b, (void*)v_t, MA, GD, GD, GTA, nullptr, nullptr);

  // 4) flash cross-attention
  flash_attn_kernel<<<dim3(GTT / 16, GB * GH), 32, 0, stream>>>(
      q_h, k_h, v_t, amask, ao, GTT, GTA);

  // 5) output projection + gated residual (f32 out)
  wmma_gemm_kernel<3><<<dim3(MT / 16, GD / 128), 256, 0, stream>>>(
      ao, wo_b, d_out, MT, GD, GD, GTT, hs, gate);
}
